// KDE_76347338654399
// MI455X (gfx1250) — matrix operand order repaired and checked by reading, not compiled, not run on hardware
//
#include <hip/hip_runtime.h>
#include <math.h>

// ---------------------------------------------------------------------------
// KDE log-density on gfx1250:  log_p[b] = log( sum_n exp(-scale * ||x_b - sv_n||^2) )
//                                         - log(N) + (D/2) log(scale/pi)
// GEMM part done with V_WMMA_F32_16X16X32_BF16 using the 3-product bf16 split
// (hi*hi + hi*lo + lo*hi) for near-fp32 accuracy at bf16 matrix throughput.
// Epilogue folded into exp2 space:  exp(-scale*d2) = 2^( nc*x2 + nc*y2 + m2nc*(x.y) )
// with nc = -scale*log2(e), m2nc = -2*nc, so each output element is fma+exp.
// ---------------------------------------------------------------------------

typedef __attribute__((ext_vector_type(16))) __bf16 v16bf;
typedef __attribute__((ext_vector_type(8)))  float  v8f;

#define B_ROWS   2048
#define N_SVS    65536
#define D_DIM    64
#define MTILES   4                      // 4 x 16 = 64 X-rows per wave (A reuse)
#define NCHUNKS  128                    // N split into 128 chunks
#define NTILES_PER_CHUNK (N_SVS / 16 / NCHUNKS)   // 32 sv tiles of 16 rows
#define BGROUPS  (B_ROWS / (16 * MTILES))         // 32

__device__ __forceinline__ void bf16_split(float f, __bf16& hi, __bf16& lo) {
    hi = (__bf16)f;
    lo = (__bf16)(f - (float)hi);
}

// One wave (32 threads) per workgroup. blockIdx.x = N chunk, blockIdx.y = B group.
__global__ __launch_bounds__(32)
void kde_wmma_kernel(const float* __restrict__ X,
                     const float* __restrict__ svs,
                     const float* __restrict__ scale_p,
                     float* __restrict__ ws /* [NCHUNKS][B_ROWS] partial sums */)
{
    const int lane = threadIdx.x;        // 0..31 (wave32)
    const int lr   = lane & 15;          // row-within-tile / column index
    const int lh   = lane >> 4;          // lane half

    const int bbase   = blockIdx.y * (16 * MTILES);
    const int n_start = blockIdx.x * NTILES_PER_CHUNK * 16;

    const float scale = *scale_p;
    const float nc   = -scale * 1.44269504088896340736f;   // -scale * log2(e)
    const float m2nc = -2.0f * nc;                          // +2*scale*log2(e)

    // ---- A operands: MTILES x (2 K-steps) of 16x32 bf16 hi/lo, ISA layout:
    //  lanes 0-15 (lh=0): elems 0..7 -> K=0..7,  elems 8..15 -> K=16..23
    //  lanes 16-31(lh=1): elems 0..7 -> K=8..15, elems 8..15 -> K=24..31
    v16bf ahi[MTILES][2], alo[MTILES][2];
    float ncx2[MTILES][8];               // nc * ||x_row||^2, row = r + 8*lh
#pragma unroll
    for (int mt = 0; mt < MTILES; ++mt) {
        const int brow = bbase + mt * 16 + lr;
#pragma unroll
        for (int kk = 0; kk < 2; ++kk) {
            const float* xrow = X + (size_t)brow * D_DIM + kk * 32;
#pragma unroll
            for (int j = 0; j < 8; ++j) {
                __bf16 h, l;
                bf16_split(xrow[8 * lh + j], h, l);
                ahi[mt][kk][j] = h; alo[mt][kk][j] = l;
                bf16_split(xrow[16 + 8 * lh + j], h, l);
                ahi[mt][kk][8 + j] = h; alo[mt][kk][8 + j] = l;
            }
        }
        // ||x||^2 for this tile's 16 rows (lanes 0-15 and 16-31 duplicate)
        float x2row = 0.f;
        const float* xr = X + (size_t)brow * D_DIM;
#pragma unroll
        for (int d = 0; d < D_DIM; ++d) { float v = xr[d]; x2row += v * v; }
        // C/D layout: reg r, lane l -> row = r + 8*(l>>4)
#pragma unroll
        for (int r = 0; r < 8; ++r)
            ncx2[mt][r] = nc * __shfl(x2row, 8 * lh + r, 32);
    }

    float accr[MTILES][8];
#pragma unroll
    for (int mt = 0; mt < MTILES; ++mt)
#pragma unroll
        for (int r = 0; r < 8; ++r) accr[mt][r] = 0.f;

    // ---- stream sv tiles (16 rows each) through WMMA
    for (int t = 0; t < NTILES_PER_CHUNK; ++t) {
        const int n0 = n_start + t * 16;
        const float* srow = svs + (size_t)(n0 + lr) * D_DIM;

        // B operand 32x16 bf16: lane = column n0+lr; same K layout as A (lh selects 8-wide K blocks).
        v16bf bhi[2], blo[2];
        float psq = 0.f;
#pragma unroll
        for (int kk = 0; kk < 2; ++kk) {
            const float* p = srow + kk * 32 + 8 * lh;
#pragma unroll
            for (int j = 0; j < 16; ++j) {
                float f = p[(j & 7) + ((j >> 3) << 4)];
                psq += f * f;
                __bf16 h, l;
                bf16_split(f, h, l);
                bhi[kk][j] = h; blo[kk][j] = l;
            }
        }
        // full nc*||sv||^2 for this lane's column = own half + other half
        const float ncy2 = nc * (psq + __shfl_xor(psq, 16, 32));

#pragma unroll
        for (int mt = 0; mt < MTILES; ++mt) {
            v8f c = {};
#pragma unroll
            for (int kk = 0; kk < 2; ++kk) {
                c = __builtin_amdgcn_wmma_f32_16x16x32_bf16(
                        false, ahi[mt][kk], false, bhi[kk], (short)0, c, false, false);
                c = __builtin_amdgcn_wmma_f32_16x16x32_bf16(
                        false, ahi[mt][kk], false, blo[kk], (short)0, c, false, false);
                c = __builtin_amdgcn_wmma_f32_16x16x32_bf16(
                        false, alo[mt][kk], false, bhi[kk], (short)0, c, false, false);
            }
            // exp(-scale*d2) = exp2( ncx2 + ncy2 + m2nc*(x.y) ): fma + exp per elem
#pragma unroll
            for (int r = 0; r < 8; ++r) {
                float base = ncx2[mt][r] + ncy2;
                accr[mt][r] += __builtin_amdgcn_exp2f(fmaf(m2nc, c[r], base));
            }
        }
    }

    // ---- reduce over the 16 columns in each lane half; write partial sums
    float* wchunk = ws + (size_t)blockIdx.x * B_ROWS;
#pragma unroll
    for (int mt = 0; mt < MTILES; ++mt) {
#pragma unroll
        for (int r = 0; r < 8; ++r) {
            float s = accr[mt][r];
            s += __shfl_xor(s, 1, 32);
            s += __shfl_xor(s, 2, 32);
            s += __shfl_xor(s, 4, 32);
            s += __shfl_xor(s, 8, 32);   // all lanes in a half now hold the half-sum
            if (lr == 0)                  // lanes 0 and 16: rows r and r+8
                wchunk[bbase + mt * 16 + 8 * lh + r] = s;
        }
    }
}

// Deterministic finalize: fixed-order reduction over chunks + log terms.
__global__ void kde_final_kernel(const float* __restrict__ ws,
                                 const float* __restrict__ scale_p,
                                 float* __restrict__ out)
{
    const int b = blockIdx.x * blockDim.x + threadIdx.x;
    if (b >= B_ROWS) return;
    float s = 0.f;
    for (int c = 0; c < NCHUNKS; ++c)
        s += ws[(size_t)c * B_ROWS + b];
    const float scale = *scale_p;
    out[b] = logf(s) - logf((float)N_SVS)
           + (D_DIM * 0.5f) * logf(scale / 3.14159265358979323846f);
}

extern "C" void kernel_launch(void* const* d_in, const int* in_sizes, int n_in,
                              void* d_out, int out_size, void* d_ws, size_t ws_size,
                              hipStream_t stream) {
    (void)in_sizes; (void)n_in; (void)out_size; (void)ws_size;
    const float* X     = (const float*)d_in[0];
    const float* svs   = (const float*)d_in[1];
    const float* scale = (const float*)d_in[2];
    float* ws  = (float*)d_ws;    // NCHUNKS * B_ROWS floats = 1 MB
    float* out = (float*)d_out;

    dim3 grid(NCHUNKS, BGROUPS);  // 128 x 32 single-wave workgroups
    kde_wmma_kernel<<<grid, 32, 0, stream>>>(X, svs, scale, ws);
    kde_final_kernel<<<(B_ROWS + 255) / 256, 256, 0, stream>>>(ws, scale, out);
}
